// RenderNet_13494787244318
// MI455X (gfx1250) — matrix operand order repaired and checked by reading, not compiled, not run on hardware
//
#include <hip/hip_runtime.h>
#include <math.h>

// ---------------------------------------------------------------------------
// RenderNet line rasterize + 7x7 conv + minmax-normalize, collapsed to:
//   out[c,h,w] = 1 - A[h,w]/Amax,  A = correlation of line indicator with K.
// Output write (201 MB) is the roofline bottleneck (~8.6 us @ 23.3 TB/s).
// Band compute: 16x16 tiles as A_tile = U(16x154)*V(154x16) via 5 unrolled
// v_wmma_f32_16x16x32_f16 (exact: K in {0.25,0.5,0.75,1.0}, f32 accumulate).
// Fragment build is branchless: zero-padded shifted LDS tables -> every inner
// element is one ds_load with an immediate offset (no predication/bpermute).
// ---------------------------------------------------------------------------

typedef __attribute__((ext_vector_type(16))) _Float16 v16h;
typedef __attribute__((ext_vector_type(8)))  float    v8f;

#define I32_MIN ((int)0x80000000)

// Workspace layout (int units):
//   ws[0]=xmin  ws[1]=ymin  ws[2]=N  ws[3]=padding  ws[4]=imsize  ws[5]=lw
//   ws[8]      = Amax as float bits (uint)
//   ws[16 ...] = ysv[N]  (rounded line y per line point, includes +padding)
#define WS_AMAX 8
#define WS_YSV  16

__global__ void setup_k(const int* __restrict__ x0, const int* __restrict__ y0,
                        const int* __restrict__ x1, const int* __restrict__ y1,
                        const int* __restrict__ pim, const int* __restrict__ plw,
                        int* __restrict__ ws) {
  const int X0 = *x0, Y0 = *y0, X1 = *x1, Y1 = *y1;
  const int im = *pim, lw = *plw;
  const int xmin = min(X0, X1), xmax = max(X0, X1);
  const int ymin = min(Y0, Y1), ymax = max(Y0, Y1);
  const int pad  = lw - 1;
  const int N    = xmax - xmin + 1;
  if (threadIdx.x == 0) {
    ws[0] = xmin; ws[1] = ymin; ws[2] = N;
    ws[3] = pad;  ws[4] = im;   ws[5] = lw;
    ((unsigned*)ws)[WS_AMAX] = 0u;                       // reset Amax each call
  }
  // float64 math + rint() == numpy float64 round-half-even semantics
  const double slope = (double)(ymax - ymin) / (double)(xmax - xmin);
  int* ysv = ws + WS_YSV;
  for (int k = (int)threadIdx.x; k < N; k += (int)blockDim.x) {
    double xs = (double)(xmin + k + pad);                // xs already has +pad
    ysv[k] = (int)rint(xs * slope + (double)ymin) + pad;
  }
}

// Streaming fill of 1.0 — this IS the kernel's runtime (write-BW bound).
__global__ void fill_k(float* __restrict__ out, long long nvec, long long total) {
  long long i = (long long)blockIdx.x * blockDim.x + threadIdx.x;
  const long long stride = (long long)gridDim.x * blockDim.x;
  const float4 one = make_float4(1.f, 1.f, 1.f, 1.f);
  for (; i < nvec; i += stride) ((float4*)out)[i] = one;
  if (blockIdx.x == 0 && threadIdx.x == 0)
    for (long long j = nvec * 4; j < total; ++j) out[j] = 1.0f;
}

// --- Amax: direct LW-tap stencil at each candidate band pixel, atomic max ---
template <int LW>
__global__ void amax_t(const float* __restrict__ Kw, int* __restrict__ ws) {
  const int xmin = ws[0], N = ws[2], im = ws[4];
  const int PAD = LW - 1;
  const int* ysv = ws + WS_YSV;
  const int idx = blockIdx.x * blockDim.x + threadIdx.x;
  const int k = idx >> 6, off = idx & 63;
  if (k >= N || off >= LW * LW) return;
  const int kh = off / LW, kw = off % LW;
  const int h = xmin + k + PAD - kh;
  const int w = ysv[k] - kw;
  if (h < 0 || h >= im || w < 0 || w >= im) return;
  float A = 0.f;
#pragma unroll
  for (int dk = 0; dk < LW; ++dk) {
    int kp = h - xmin - PAD + dk;
    if (kp >= 0 && kp < N) {
      int dy = ysv[kp] - w;
      if (dy >= 0 && dy < LW) A += Kw[dk * LW + dy];
    }
  }
  atomicMax((unsigned*)(ws + WS_AMAX), __float_as_uint(A));  // A >= 0
}

// --- Band tiles via WMMA (one wave per 16x16 tile), LW compile-time ---------
template <int LW>
__global__ void band_t(const float* __restrict__ Kw, const int* __restrict__ ws,
                       float* __restrict__ out) {
  constexpr int PAD   = LW - 1;
  constexpr int SR    = 16 + LW - 1;            // candidate line points / group
  constexpr int INNER = SR * LW;                // matmul inner dim (154)
  constexpr int STEPS = (INNER + 31) / 32;      // 5 WMMAs
  constexpr int KOFF  = (LW * 15);              // 105: shift so index >= 0
  constexpr int K2SZ  = KOFF + 32 * STEPS + 8;  // 273 -> pad below
  constexpr int CTSZ  = ((32 * STEPS + 16 + 31) / 32) * 32;  // 192

  const int xmin = ws[0], N = ws[2], im = ws[4];
  const int* ysv = ws + WS_YSV;
  const int h0   = (int)blockIdx.x * 16;
  const int lane = (int)threadIdx.x;
  const int hl   = lane & 15;
  const bool hi  = lane >= 16;

  __shared__ _Float16 sK2[((K2SZ + 31) / 32) * 32];  // shifted, zero-padded K
  __shared__ int      sCT[CTSZ];                     // target column per t
  __shared__ int      sY[32];

  // One relevant ysv per lane; INT_MIN encodes "no line point".
  const int kp0 = h0 - xmin - PAD;
  int myY = I32_MIN;
  {
    int kp = kp0 + lane;
    if (lane < SR && kp >= 0 && kp < N) myY = ysv[kp];
  }
  sY[lane] = myY;

  // Shifted K table: sK2[KOFF + i] = Kflat[i], zero elsewhere (no bounds
  // checks needed in the hot loop: all reachable indices land in-table).
#pragma unroll
  for (int idx = lane; idx < (int)(sizeof(sK2) / sizeof(sK2[0])); idx += 32) {
    int i = idx - KOFF;
    float v = ((unsigned)i < (unsigned)(LW * LW)) ? Kw[i] : 0.f;
    sK2[idx] = (_Float16)v;
  }
  __syncthreads();

  // Target-column table: sCT[t] = ysv[t/LW] - t%LW. Invalid slots hold
  // INT_MIN - dy, which wraps to ~2^31 and can never equal a real column.
#pragma unroll
  for (int t = lane; t < CTSZ; t += 32)
    sCT[t] = sY[t / LW] - (t % LW);

  // Column window of the band for this row group (wave min/max reduce).
  int mn = (myY == I32_MIN) ? INT_MAX : myY - PAD;
  int mx = myY;                                  // INT_MIN acts as -inf
  for (int o = 16; o > 0; o >>= 1) {
    mn = min(mn, __shfl_xor(mn, o, 32));
    mx = max(mx, __shfl_xor(mx, o, 32));
  }
  __syncthreads();
  if (mn == INT_MAX) return;                     // uniform: no line here
  const int w0 = (mn & ~15) + (int)blockIdx.y * 16;
  if (w0 > mx) return;                           // uniform: right of band

  // Per-lane constant bases; every inner element is then a single LDS load
  // at (base + unroll-literal) -> ds_load with immediate offset, branchless.
  const int aBase = KOFF + (hi ? 8 : 0) - LW * hl;   // into sK2 (elements)
  const int bBase = hi ? 8 : 0;                      // into sCT (elements)
  const int wTgt  = w0 + hl;
  const _Float16 oneH = (_Float16)1.f, zeroH = (_Float16)0.f;

  v8f acc = {};
#pragma unroll
  for (int st = 0; st < STEPS; ++st) {
    v16h a, b;
#pragma unroll
    for (int j = 0; j < 16; ++j) {
      // A (ISA layout): K-index ka = 32*st + (j<8?j:j+8) + 8*hi;
      // U[hl, t] = Kflat[t - LW*hl] -> single shifted-table load.
      const int kaLo = 32 * st + (j < 8 ? j : j + 8);
      a[j] = sK2[aBase + kaLo];
      // B: K-index kb = 32*st + (j<8?j:j+8) + 8*hi; hit iff target column == w0+hl.
      const int kbLo = 32 * st + (j < 8 ? j : j + 8);
      int ct = sCT[bBase + kbLo];
      b[j] = (ct == wTgt) ? oneH : zeroH;
    }
    acc = __builtin_amdgcn_wmma_f32_16x16x32_f16(
        false, a, false, b, (short)0, acc, false, false);
  }

  const float amax = __uint_as_float(((const unsigned*)ws)[WS_AMAX]);
  const float inv  = (amax > 0.f) ? 1.0f / amax : 0.f;
  const size_t plane = (size_t)im * (size_t)im;
#pragma unroll
  for (int q = 0; q < 8; ++q) {                  // C/D: M = q + 8*hi, N = hl
    int hh = h0 + q + (hi ? 8 : 0);
    int ww = w0 + hl;
    if (hh < im && ww >= 0 && ww < im) {
      float v = 1.0f - acc[q] * inv;
      size_t p = (size_t)hh * (size_t)im + (size_t)ww;
      out[p] = v; out[p + plane] = v; out[p + 2 * plane] = v;
    }
  }
}

// --- Generic fallbacks (lw != 7): direct stencil, per-candidate writes ------
__global__ void amax_g(const float* __restrict__ Kw, int* __restrict__ ws) {
  const int xmin = ws[0], N = ws[2], pad = ws[3], im = ws[4], lw = ws[5];
  const int* ysv = ws + WS_YSV;
  const int idx = blockIdx.x * blockDim.x + threadIdx.x;
  const int k = idx >> 6, off = idx & 63;
  if (k >= N || off >= lw * lw) return;
  const int h = xmin + k + pad - off / lw;
  const int w = ysv[k] - off % lw;
  if (h < 0 || h >= im || w < 0 || w >= im) return;
  float A = 0.f;
  for (int dk = 0; dk < lw; ++dk) {
    int kp = h - xmin - pad + dk;
    if (kp >= 0 && kp < N) {
      int dy = ysv[kp] - w;
      if (dy >= 0 && dy < lw) A += Kw[dk * lw + dy];
    }
  }
  atomicMax((unsigned*)(ws + WS_AMAX), __float_as_uint(A));
}

__global__ void band_g(const float* __restrict__ Kw, const int* __restrict__ ws,
                       float* __restrict__ out) {
  const int xmin = ws[0], N = ws[2], pad = ws[3], im = ws[4], lw = ws[5];
  const int* ysv = ws + WS_YSV;
  const int idx = blockIdx.x * blockDim.x + threadIdx.x;
  const int k = idx >> 6, off = idx & 63;
  if (k >= N || off >= lw * lw) return;
  const int h = xmin + k + pad - off / lw;
  const int w = ysv[k] - off % lw;
  if (h < 0 || h >= im || w < 0 || w >= im) return;
  float A = 0.f;
  for (int dk = 0; dk < lw; ++dk) {
    int kp = h - xmin - pad + dk;
    if (kp >= 0 && kp < N) {
      int dy = ysv[kp] - w;
      if (dy >= 0 && dy < lw) A += Kw[dk * lw + dy];
    }
  }
  const float amax = __uint_as_float(((const unsigned*)ws)[WS_AMAX]);
  const float v = 1.0f - A * ((amax > 0.f) ? 1.0f / amax : 0.f);
  const size_t plane = (size_t)im * (size_t)im;
  size_t p = (size_t)h * (size_t)im + (size_t)w;
  out[p] = v; out[p + plane] = v; out[p + 2 * plane] = v;
}

extern "C" void kernel_launch(void* const* d_in, const int* in_sizes, int n_in,
                              void* d_out, int out_size, void* d_ws, size_t ws_size,
                              hipStream_t stream) {
  const float* Kw  = (const float*)d_in[0];   // (3,3,7,7), all 9 copies equal
  const int*   x0  = (const int*)d_in[1];
  const int*   y0  = (const int*)d_in[2];
  const int*   x1  = (const int*)d_in[3];
  const int*   y1  = (const int*)d_in[4];
  const int*   pim = (const int*)d_in[5];
  const int*   plw = (const int*)d_in[6];
  float* out = (float*)d_out;
  int*   ws  = (int*)d_ws;

  // Host-side sizing (exact: out_size = 3*im^2, in_sizes[0] = 9*lw^2).
  const int im = (int)(sqrt((double)out_size / 3.0) + 0.5);
  const int lw = (int)(sqrt((double)in_sizes[0] / 9.0) + 0.5);

  setup_k<<<1, 256, 0, stream>>>(x0, y0, x1, y1, pim, plw, ws);

  const long long total = (long long)out_size;
  const long long nvec  = total / 4;
  fill_k<<<16384, 256, 0, stream>>>(out, nvec, total);

  const int Nmax = im + lw;                              // >= actual N
  const int athreads = Nmax * 64;                        // 64 slots per point
  const int ablocks  = (athreads + 255) / 256;

  if (lw == 7) {
    amax_t<7><<<ablocks, 256, 0, stream>>>(Kw, ws);
    dim3 bg((im + 15) / 16, 8);                          // extra tiles early-exit
    band_t<7><<<bg, 32, 0, stream>>>(Kw, ws, out);
  } else {
    amax_g<<<ablocks, 256, 0, stream>>>(Kw, ws);
    band_g<<<ablocks, 256, 0, stream>>>(Kw, ws, out);
  }
}
